// SwinTransformerBlock_60455959658929
// MI455X (gfx1250) — compile-verified, matrix operand order checked by reading, not run on hardware
//
#include <hip/hip_runtime.h>
#include <math.h>

// ---------------- problem constants ----------------
#define BB      32
#define HH      56
#define WW_     56
#define CC      128
#define HEADS   4
#define DH      32          // head dim
#define WS      7
#define SHIFT   3
#define PP      10          // prompts
#define NWIN    49
#define NPAD    64          // padded tokens per window (59 -> 64)
#define NW      64          // windows per image (8x8)
#define BWIN    (BB*NW)     // 2048 window-batches
#define MROWS   (BWIN*NPAD) // 131072 token rows (padded)
#define HID     512
#define HW      (HH*WW_)    // 3136
#define MX      (BB*HW)     // 100352 spatial rows

// workspace byte offsets
#define OFF_WQKV  ((size_t)0)
#define OFF_WPROJ ((size_t)98304)
#define OFF_WFC1  ((size_t)131072)
#define OFF_WFC2  ((size_t)262144)
#define OFF_A1    ((size_t)4  << 20)   // t / o / ln2-out (<= 33.6 MB)
#define OFF_A2    ((size_t)40 << 20)   // qkv planar (100.7 MB) then x1 fp32 (51.4 MB)
#define OFF_A4    ((size_t)144 << 20)  // fc1 output bf16 (102.8 MB)

typedef __bf16 v16bf __attribute__((ext_vector_type(16)));
typedef float  v8f   __attribute__((ext_vector_type(8)));

__device__ __forceinline__ v8f zero8() {
  v8f z = {0.f,0.f,0.f,0.f,0.f,0.f,0.f,0.f};
  return z;
}
__device__ __forceinline__ v8f wmma_bf16(v16bf a, v16bf b, v8f c) {
  return __builtin_amdgcn_wmma_f32_16x16x32_bf16(false, a, false, b, (short)0, c, false, false);
}

// A fragment: 16x32 bf16 tile at (row0,col0), row-major LDS, leading dim ld.
// Per lane: rows row0+(lane&15); K split per lane-half. Two contiguous 16B runs.
__device__ __forceinline__ v16bf load_frag_A(const __bf16* lds, int row0, int col0, int ld) {
  int lane = threadIdx.x & 31;
  int r = row0 + (lane & 15);
  int g = (lane >> 4) & 1;
  v16bf f;
#pragma unroll
  for (int p = 0; p < 8; ++p) {
    int k = (p < 4) ? (col0 + p*2 + g*8) : (col0 + 16 + (p-4)*2 + g*8);
    f[2*p]   = lds[r*ld + k];
    f[2*p+1] = lds[r*ld + k + 1];
  }
  return f;
}
// B fragment from n-major storage: element (k,n) = lds[(n0+n)*ld + kbase + k].
// Per lane: one n-row, two contiguous 16B K-runs -> ds_load_b128 pairs.
__device__ __forceinline__ v16bf load_frag_Bt(const __bf16* lds, int n0, int kbase, int ld) {
  int lane = threadIdx.x & 31;
  int n = n0 + (lane & 15);
  int g = (lane >> 4) & 1;
  v16bf f;
#pragma unroll
  for (int p = 0; p < 8; ++p) {
    int k = (p < 4) ? (p*2 + g*8) : (16 + (p-4)*2 + g*8);
    f[2*p]   = lds[n*ld + kbase + k];
    f[2*p+1] = lds[n*ld + kbase + k + 1];
  }
  return f;
}

// ---------------- LoRA weight folding: Weff^T = (W + la@lb)^T (bf16, N x K) ----
__global__ void prep_w(const float* __restrict__ w, const float* __restrict__ la,
                       const float* __restrict__ lb, __bf16* __restrict__ dst,
                       int K, int Nc) {
  int idx = blockIdx.x * 256 + threadIdx.x;
  if (idx >= K * Nc) return;
  int i = idx / Nc, j = idx - i * Nc;      // i: K index, j: N index
  float acc = w[idx];
#pragma unroll
  for (int r = 0; r < 4; ++r) acc += la[i*4 + r] * lb[r*Nc + j];
  dst[(size_t)j * K + i] = (__bf16)acc;    // transposed store
}

// ---------------- block-of-128 row reduction ----------------
__device__ __forceinline__ float block128_sum(float v, float* red) {
#pragma unroll
  for (int m = 16; m >= 1; m >>= 1) v += __shfl_xor(v, m, 32);
  int wid = threadIdx.x >> 5;
  if ((threadIdx.x & 31) == 0) red[wid] = v;
  __syncthreads();
  float s = red[0] + red[1] + red[2] + red[3];
  __syncthreads();
  return s;
}

// ---------------- LN1 + roll(-3,-3) + window partition + prompts -> t (bf16) ----
__global__ __launch_bounds__(128) void build_t(const float* __restrict__ x,
                                               const float* __restrict__ sp,
                                               const float* __restrict__ g,
                                               const float* __restrict__ bta,
                                               __bf16* __restrict__ t) {
  __shared__ float red[4];
  int row = blockIdx.x;          // 0..131071
  int c = threadIdx.x;           // channel
  int b_ = row >> 6, n = row & 63;
  int b = b_ >> 6, w = b_ & 63;
  size_t drow = (size_t)row * CC;
  if (n >= PP + NWIN) { t[drow + c] = (__bf16)0.f; return; }       // pad rows -> 0
  if (n < PP) { t[drow + c] = (__bf16)sp[((size_t)b*PP + n)*CC + c]; return; }
  int a  = (n - PP) / WS, bb = (n - PP) % WS;
  int wh = w >> 3,  ww = w & 7;
  int r  = ((wh*WS + a)  + SHIFT) % HH;    // roll by -SHIFT: h[i] = x[i+SHIFT]
  int cc = ((ww*WS + bb) + SHIFT) % WW_;
  float xv = x[((size_t)b*HW + r*WW_ + cc)*CC + c];
  float mean = block128_sum(xv, red) * (1.f/128.f);
  float dv = xv - mean;
  float var = block128_sum(dv*dv, red) * (1.f/128.f);
  float y = dv * rsqrtf(var + 1e-5f) * g[c] + bta[c];
  t[drow + c] = (__bf16)y;
}

// ---------------- LN2 over x1 -> bf16 ----------------
__global__ __launch_bounds__(128) void ln2_kernel(const float* __restrict__ x1,
                                                  const float* __restrict__ g,
                                                  const float* __restrict__ bta,
                                                  __bf16* __restrict__ out) {
  __shared__ float red[4];
  size_t row = blockIdx.x;
  int c = threadIdx.x;
  float xv = x1[row*CC + c];
  float mean = block128_sum(xv, red) * (1.f/128.f);
  float dv = xv - mean;
  float var = block128_sum(dv*dv, red) * (1.f/128.f);
  out[row*CC + c] = (__bf16)(dv * rsqrtf(var + 1e-5f) * g[c] + bta[c]);
}

// ---------------- generic bf16 WMMA GEMM (W pre-transposed) + fused epilogues ---
#define MODE_QKV  0
#define MODE_PROJ 1
#define MODE_GELU 2
#define MODE_OUT  3
#define QKV_PLANE ((size_t)BWIN*HEADS*NPAD*DH)   // 16,777,216 elements per q/k/v

__global__ __launch_bounds__(256) void wmma_gemm(
    const __bf16* __restrict__ A, const __bf16* __restrict__ Wt, // Wt: N x K
    const float* __restrict__ bias, int M, int N, int K, int mode,
    __bf16* __restrict__ outb, float* __restrict__ outf,
    const float* __restrict__ aux) {
  __shared__ __align__(16) __bf16 As [64  * 48];   // 64x32 (m,k),  ld=48 (96B rows)
  __shared__ __align__(16) __bf16 Bst[128 * 48];   // 128x32 (n,k), ld=48 (96B rows)
  int tid = threadIdx.x;
  int rowBase = blockIdx.y * 64;
  int colBase = blockIdx.x * 128;
  int wave = tid >> 5, lane = tid & 31;
  int wr = wave >> 2, wc = wave & 3;               // 2x4 waves of 32x32 outputs
  v8f acc[2][2];
  acc[0][0] = zero8(); acc[0][1] = zero8(); acc[1][0] = zero8(); acc[1][1] = zero8();

  for (int k0 = 0; k0 < K; k0 += 32) {
    { // stage A 64x32 (8 bf16 / thread, one 16B vector)
      int r = tid >> 2, cs = (tid & 3) * 8;
      const uint4* src = (const uint4*)(A + (size_t)(rowBase + r) * K + k0 + cs);
      *(uint4*)(As + r*48 + cs) = *src;
      if (k0 + 32 < K)  // prefetch next A k-slice (global_prefetch_b8)
        __builtin_prefetch(A + (size_t)(rowBase + r) * K + k0 + 32 + cs, 0, 1);
    }
    { // stage B 128x32 n-major (16 bf16 / thread, two 16B vectors)
      int n = tid >> 1, ks = (tid & 1) * 16;
      const uint4* src = (const uint4*)(Wt + (size_t)(colBase + n) * K + k0 + ks);
      uint4 u0 = src[0], u1 = src[1];
      *(uint4*)(Bst + n*48 + ks)     = u0;
      *(uint4*)(Bst + n*48 + ks + 8) = u1;
    }
    __syncthreads();
    v16bf a0 = load_frag_A (As,  wr*32 + 0,  0, 48);
    v16bf a1 = load_frag_A (As,  wr*32 + 16, 0, 48);
    v16bf b0 = load_frag_Bt(Bst, wc*32 + 0,  0, 48);
    v16bf b1 = load_frag_Bt(Bst, wc*32 + 16, 0, 48);
    acc[0][0] = wmma_bf16(a0, b0, acc[0][0]);
    acc[0][1] = wmma_bf16(a0, b1, acc[0][1]);
    acc[1][0] = wmma_bf16(a1, b0, acc[1][0]);
    acc[1][1] = wmma_bf16(a1, b1, acc[1][1]);
    __syncthreads();
  }

  int lg = lane >> 4, ln = lane & 15;
#pragma unroll
  for (int ti = 0; ti < 2; ++ti)
#pragma unroll
    for (int tj = 0; tj < 2; ++tj)
#pragma unroll
      for (int e = 0; e < 8; ++e) {
        int row = rowBase + wr*32 + ti*16 + e + 8*lg;
        int col = colBase + wc*32 + tj*16 + ln;
        float val = acc[ti][tj][e];
        if (mode == MODE_QKV) {
          float v = val + bias[col];
          int which = col >> 7, hc = col & 127;
          int h = hc >> 5, dc = hc & 31;
          int b_ = row >> 6, n = row & 63;
          size_t dst;
          if (which == 2)  // V stored transposed: [b_,h][dc][token]
            dst = 2*QKV_PLANE + ((size_t)(b_*HEADS + h)*DH + dc)*NPAD + n;
          else
            dst = (size_t)which * QKV_PLANE +
                  ((size_t)((b_*HEADS + h)*NPAD + n)) * DH + dc;
          outb[dst] = (__bf16)v;
        } else if (mode == MODE_PROJ) {
          int n = row & 63;
          if (n >= PP && n < PP + NWIN) {
            float v = val + bias[col];
            int b_ = row >> 6, b = b_ >> 6, w = b_ & 63;
            int a = (n - PP) / WS, bb = (n - PP) % WS;
            int r  = (w >> 3)*WS + a;
            int c2 = (w & 7)*WS + bb;
            int rr = (r + SHIFT) % HH;          // roll back by +SHIFT
            int cc = (c2 + SHIFT) % WW_;
            size_t idx = ((size_t)b*HW + rr*WW_ + cc)*CC + col;
            outf[idx] = aux[idx] + v;           // shortcut + h
          }
        } else if (mode == MODE_GELU) {
          float v = val + bias[col];
          v = 0.5f * v * (1.f + erff(v * 0.70710678118654752f));
          outb[(size_t)row * N + col] = (__bf16)v;
        } else { // MODE_OUT
          float v = val + bias[col];
          size_t idx = (size_t)row * CC + col;
          outf[idx] = aux[idx] + v;             // x1 + mlp
        }
      }
}

// ---------------- fused windowed attention (1 wave per head) ----------------
__device__ __forceinline__ int regid(int r) { return r < 49 ? 0 : (r < 53 ? 1 : 2); }

__global__ __launch_bounds__(128) void attn_kernel(const __bf16* __restrict__ qkv,
                                                   const float* __restrict__ rpb,
                                                   __bf16* __restrict__ o) {
  __shared__ __align__(16) __bf16 Qs[4][NPAD*DH];   // [token][dh]
  __shared__ __align__(16) __bf16 Ks[4][NPAD*DH];   // [token][dh] (consumed as K^T)
  __shared__ __align__(16) __bf16 Vs[4][DH*NPAD];   // [dh][token] (transposed in HBM)
  __shared__ __align__(16) __bf16 Ps[4][16*NPAD];   // current P row-tile
  int b_ = blockIdx.x;
  int h = threadIdx.x >> 5, lane = threadIdx.x & 31;
  size_t base = ((size_t)(b_*HEADS + h)) * (NPAD*DH);
  const uint4* qg = (const uint4*)(qkv + base);
  const uint4* kg = (const uint4*)(qkv + QKV_PLANE + base);
  const uint4* vg = (const uint4*)(qkv + 2*QKV_PLANE + base);
  uint4* ql = (uint4*)Qs[h]; uint4* kl = (uint4*)Ks[h]; uint4* vl = (uint4*)Vs[h];
#pragma unroll
  for (int u = 0; u < 8; ++u) {
    ql[lane + 32*u] = qg[lane + 32*u];
    kl[lane + 32*u] = kg[lane + 32*u];
    vl[lane + 32*u] = vg[lane + 32*u];
  }
  __syncthreads();

  int lg = lane >> 4, ln = lane & 15;
  int w = b_ & 63, wh = w >> 3, ww = w & 7;
  const float scale = 0.17677669529663687f;    // 1/sqrt(32)

  for (int i = 0; i < 4; ++i) {                // 16-row tiles of S
    v16bf qa = load_frag_A(Qs[h], i*16, 0, DH);
    v8f s[4];
#pragma unroll
    for (int j = 0; j < 4; ++j) {
      v16bf kb = load_frag_Bt(Ks[h], j*16, 0, DH);   // K^T tile, contiguous runs
      s[j] = wmma_bf16(qa, kb, zero8());
    }
    // scale + rel-pos bias + shift mask + pad mask
#pragma unroll
    for (int j = 0; j < 4; ++j) {
      int m = j*16 + ln;
#pragma unroll
      for (int e = 0; e < 8; ++e) {
        int n = i*16 + e + 8*lg;
        float v = s[j][e] * scale;
        if (m >= PP + NWIN) v = -1e30f;         // padded key columns
        else if (n >= PP && m >= PP) {
          int ii = n - PP, jj = m - PP;
          int ih = ii / WS, iw = ii % WS, jh = jj / WS, jw = jj % WS;
          int rpi = (ih - jh + WS - 1) * (2*WS - 1) + (iw - jw + WS - 1);
          v += rpb[rpi * HEADS + h];
          int idi = regid(wh*WS + ih)*3 + regid(ww*WS + iw);
          int idj = regid(wh*WS + jh)*3 + regid(ww*WS + jw);
          if (idi != idj) v -= 100.f;
        }
        s[j][e] = v;
      }
    }
    // masked softmax: each row lives in one 16-lane half across the 4 j-tiles
#pragma unroll
    for (int e = 0; e < 8; ++e) {
      float mx = fmaxf(fmaxf(s[0][e], s[1][e]), fmaxf(s[2][e], s[3][e]));
#pragma unroll
      for (int msk = 1; msk < 16; msk <<= 1) mx = fmaxf(mx, __shfl_xor(mx, msk, 32));
      float pj[4], sm = 0.f;
#pragma unroll
      for (int j = 0; j < 4; ++j) { pj[j] = __expf(s[j][e] - mx); sm += pj[j]; }
#pragma unroll
      for (int msk = 1; msk < 16; msk <<= 1) sm += __shfl_xor(sm, msk, 32);
      float inv = 1.f / sm;
#pragma unroll
      for (int j = 0; j < 4; ++j) s[j][e] = pj[j] * inv;
    }
    // P tile (16x64) -> LDS bf16, then A-frags for O = P @ V
#pragma unroll
    for (int j = 0; j < 4; ++j)
#pragma unroll
      for (int e = 0; e < 8; ++e)
        Ps[h][(e + 8*lg)*NPAD + j*16 + ln] = (__bf16)s[j][e];
    asm volatile("s_wait_dscnt 0" ::: "memory");

    v16bf pa0 = load_frag_A(Ps[h], 0, 0,  NPAD);
    v16bf pa1 = load_frag_A(Ps[h], 0, 32, NPAD);
#pragma unroll
    for (int ot = 0; ot < 2; ++ot) {
      // V^T in LDS: element (k=token, n=dh) = Vs[n*NPAD + k] -> contiguous runs
      v16bf vb0 = load_frag_Bt(Vs[h], ot*16, 0,  NPAD);
      v16bf vb1 = load_frag_Bt(Vs[h], ot*16, 32, NPAD);
      v8f oacc = wmma_bf16(pa0, vb0, zero8());
      oacc = wmma_bf16(pa1, vb1, oacc);
#pragma unroll
      for (int e = 0; e < 8; ++e) {
        int n = i*16 + e + 8*lg;
        o[((size_t)(b_*NPAD + n))*CC + h*DH + ot*16 + ln] = (__bf16)oacc[e];
      }
    }
  }
}

// ---------------- launcher ----------------
extern "C" void kernel_launch(void* const* d_in, const int* in_sizes, int n_in,
                              void* d_out, int out_size, void* d_ws, size_t ws_size,
                              hipStream_t stream) {
  const float* x       = (const float*)d_in[0];
  const float* sp      = (const float*)d_in[1];
  const float* n1g     = (const float*)d_in[2];
  const float* n1b     = (const float*)d_in[3];
  const float* qkv_w   = (const float*)d_in[4];
  const float* qkv_b   = (const float*)d_in[5];
  const float* qkv_la  = (const float*)d_in[6];
  const float* qkv_lb  = (const float*)d_in[7];
  const float* rpb     = (const float*)d_in[8];
  const float* proj_w  = (const float*)d_in[9];
  const float* proj_b  = (const float*)d_in[10];
  const float* proj_la = (const float*)d_in[11];
  const float* proj_lb = (const float*)d_in[12];
  const float* n2g     = (const float*)d_in[13];
  const float* n2b     = (const float*)d_in[14];
  const float* fc1_w   = (const float*)d_in[15];
  const float* fc1_b   = (const float*)d_in[16];
  const float* fc1_la  = (const float*)d_in[17];
  const float* fc1_lb  = (const float*)d_in[18];
  const float* fc2_w   = (const float*)d_in[19];
  const float* fc2_b   = (const float*)d_in[20];
  const float* fc2_la  = (const float*)d_in[21];
  const float* fc2_lb  = (const float*)d_in[22];

  char* ws = (char*)d_ws;
  __bf16* Wqkv  = (__bf16*)(ws + OFF_WQKV);
  __bf16* Wproj = (__bf16*)(ws + OFF_WPROJ);
  __bf16* Wfc1  = (__bf16*)(ws + OFF_WFC1);
  __bf16* Wfc2  = (__bf16*)(ws + OFF_WFC2);
  __bf16* tbf   = (__bf16*)(ws + OFF_A1);   // tokens -> later o -> later ln2 out
  __bf16* qkvb  = (__bf16*)(ws + OFF_A2);   // planar q/k/v (v transposed)
  float*  x1    = (float*)(ws + OFF_A2);    // reuses qkv arena after attention
  __bf16* obf   = (__bf16*)(ws + OFF_A1);
  __bf16* mbf   = (__bf16*)(ws + OFF_A1);
  __bf16* h1    = (__bf16*)(ws + OFF_A4);

  // fold LoRA into weights (bf16, transposed N x K)
  prep_w<<<(128*384 + 255)/256, 256, 0, stream>>>(qkv_w,  qkv_la,  qkv_lb,  Wqkv,  128, 384);
  prep_w<<<(128*128 + 255)/256, 256, 0, stream>>>(proj_w, proj_la, proj_lb, Wproj, 128, 128);
  prep_w<<<(128*512 + 255)/256, 256, 0, stream>>>(fc1_w,  fc1_la,  fc1_lb,  Wfc1,  128, 512);
  prep_w<<<(512*128 + 255)/256, 256, 0, stream>>>(fc2_w,  fc2_la,  fc2_lb,  Wfc2,  512, 128);

  // LN1 + shift + window partition + prompts
  build_t<<<MROWS, 128, 0, stream>>>(x, sp, n1g, n1b, tbf);

  // QKV GEMM: (131072 x 384) = t(.,128) @ Wqkv
  wmma_gemm<<<dim3(3, MROWS/64), 256, 0, stream>>>(tbf, Wqkv, qkv_b,
      MROWS, 384, 128, MODE_QKV, qkvb, nullptr, nullptr);

  // fused attention per (window, head)
  attn_kernel<<<BWIN, 128, 0, stream>>>(qkvb, rpb, obf);

  // proj GEMM + window reverse + roll back + residual -> x1
  wmma_gemm<<<dim3(1, MROWS/64), 256, 0, stream>>>(obf, Wproj, proj_b,
      MROWS, 128, 128, MODE_PROJ, nullptr, x1, x);

  // LN2 -> bf16
  ln2_kernel<<<MX, 128, 0, stream>>>(x1, n2g, n2b, mbf);

  // fc1 + GELU -> bf16
  wmma_gemm<<<dim3(4, MX/64), 256, 0, stream>>>(mbf, Wfc1, fc1_b,
      MX, 512, 128, MODE_GELU, h1, nullptr, nullptr);

  // fc2 + residual -> out
  wmma_gemm<<<dim3(1, MX/64), 256, 0, stream>>>(h1, Wfc2, fc2_b,
      MX, 128, 512, MODE_OUT, nullptr, (float*)d_out, x1);
}
